// TransformerMemoryUpdater_34376918238039
// MI455X (gfx1250) — matrix operand order repaired and checked by reading, not compiled, not run on hardware
//
#include <hip/hip_runtime.h>

// ---------------- CDNA5 WMMA types ----------------
typedef __attribute__((ext_vector_type(16))) __bf16   v16bf;
typedef __attribute__((ext_vector_type(8)))  float    v8f;
typedef __attribute__((ext_vector_type(4)))  unsigned u32x4;
typedef __attribute__((ext_vector_type(2)))  unsigned u32x2;
typedef __attribute__((ext_vector_type(4)))  float    f32x4;
typedef __attribute__((ext_vector_type(2)))  float    f32x2;

// ---------------- problem constants ----------------
#define HEADS   2
#define MAILS   10
#define DMODEL  100
#define DIN     100
#define DTENC   100
#define KKV     224          // 200 padded to 7*32
#define KQ      128          // 100 padded to 4*32
#define DP      112          // 100 padded to 7*16 (output cols)
#define NPB     16           // nodes per block
#define ROWS    (NPB*MAILS)  // 160 mail rows per block
#define THREADS 224          // 7 waves; wave w owns column tile w

// workspace layout (bf16 elements): transposed, padded weights
#define WS_WQT  0
#define WS_WKT  (112*128)
#define WS_WVT  (WS_WKT + 112*224)
#define WS_MLPT (WS_WVT + 112*224)
#define WS_TOTAL (WS_MLPT + 112*128)   // 78848 bf16 = 157696 bytes

#define SMEM_BYTES (ROWS*KKV*2 + NPB*KQ*2 + NPB*DP*4 + 2*ROWS*DP*4 + \
                    NPB*MAILS*HEADS*4 + NPB*DMODEL*4 + 2*NPB*4)     // 234112 B

// ---------------- helpers ----------------
__device__ __forceinline__ __bf16 f2bf(float f) {
  union { float f; unsigned u; } in; in.f = f;
  unsigned u = in.u;
  u += 0x7FFFu + ((u >> 16) & 1u);              // round-to-nearest-even
  union { unsigned short s; __bf16 b; } out;
  out.s = (unsigned short)(u >> 16);
  return out.b;
}

// pack two floats into one dword of bf16 (lo = a, hi = b), RNE
__device__ __forceinline__ unsigned pack2bf(float a, float b) {
  union { float f; unsigned u; } x, y; x.f = a; y.f = b;
  unsigned ua = x.u; ua += 0x7FFFu + ((ua >> 16) & 1u);
  unsigned ub = y.u; ub += 0x7FFFu + ((ub >> 16) & 1u);
  return (ua >> 16) | (ub & 0xFFFF0000u);
}

union FragU { v16bf v; u32x4 u[2]; };

// A fragment (16x32 bf16) from LDS. K order used here (same as B): lane&15 = M row,
// lane half selects K {0..15} vs {16..31}; 16 contiguous bf16
// -> two 16B LDS loads per lane.
__device__ __forceinline__ v16bf load_a_lds(const __bf16* base, int ld, int lane) {
  const int hl = lane >> 4;
  const int mr = lane & 15;
  const __bf16* p = base + mr * ld + hl * 16;
  FragU f;
  f.u[0] = *(const u32x4*)(p);       // K = hl*16 + 0..7
  f.u[1] = *(const u32x4*)(p + 8);   // K = hl*16 + 8..15
  return f.v;
}

// B fragment (32x16 bf16): lane&15 = N col, K = (lane>>4)*16 + 0..15,
// contiguous when the weight is stored K-major (pre-transposed in ws).
__device__ __forceinline__ v16bf load_b16(const __bf16* p) {
  FragU f;
  f.u[0] = *(const u32x4*)(p);
  f.u[1] = *(const u32x4*)(p + 8);
  return f.v;
}

__device__ __forceinline__ v8f wmma_bf16(v16bf a, v16bf b, v8f c) {
  return __builtin_amdgcn_wmma_f32_16x16x32_bf16(
      /*neg_a=*/false, a, /*neg_b=*/false, b,
      /*c_mod=*/(short)0, c, /*reuse_a=*/false, /*reuse_b=*/false);
}

// ---------------- weight prep: f32 -> transposed/padded bf16 in ws ----------------
__global__ void prep_weights_kernel(const float* __restrict__ wq,
                                    const float* __restrict__ wk,
                                    const float* __restrict__ wv,
                                    const float* __restrict__ mw,
                                    __bf16* __restrict__ ws) {
  int i = blockIdx.x * blockDim.x + threadIdx.x;
  if (i < 112*128) {                                   // wqT[c][k]
    int c = i / 128, k = i - c*128;
    ws[WS_WQT + i] = f2bf((c < 100 && k < 100) ? wq[k*100 + c] : 0.f);
  } else if (i < 112*128 + 112*224) {                  // wkT[c][k]
    int j = i - 112*128; int c = j / 224, k = j - c*224;
    ws[WS_WKT + j] = f2bf((c < 100 && k < 200) ? wk[k*100 + c] : 0.f);
  } else if (i < 112*128 + 2*112*224) {                // wvT[c][k]
    int j = i - (112*128 + 112*224); int c = j / 224, k = j - c*224;
    ws[WS_WVT + j] = f2bf((c < 100 && k < 200) ? wv[k*100 + c] : 0.f);
  } else if (i < 2*112*128 + 2*112*224) {              // mlpT[c][k]
    int j = i - (112*128 + 2*112*224); int c = j / 128, k = j - c*128;
    ws[WS_MLPT + j] = f2bf((c < 100 && k < 100) ? mw[k*100 + c] : 0.f);
  }
}

// ---------------- fused transformer memory updater ----------------
__global__ void __launch_bounds__(THREADS)
tmu_fused_kernel(const float* __restrict__ mem,
                 const float* __restrict__ mem_input,
                 const float* __restrict__ ts,
                 const float* __restrict__ mail_ts,
                 const float* __restrict__ bq,
                 const float* __restrict__ bk,
                 const float* __restrict__ bv,
                 const float* __restrict__ ln_g,
                 const float* __restrict__ ln_b,
                 const float* __restrict__ mlp_b,
                 const float* __restrict__ te_w,
                 const float* __restrict__ te_b,
                 const __bf16* __restrict__ ws,
                 float* __restrict__ out,
                 int n)
{
  extern __shared__ __align__(16) char smem_raw[];
  __bf16* sMails = (__bf16*)smem_raw;                  // [ROWS][KKV] bf16
  __bf16* sA     = sMails + ROWS*KKV;                  // [NPB][KQ]  bf16 (mem, later LN(rst))
  float*  sQ     = (float*)(sA + NPB*KQ);              // [NPB][DP]
  float*  sK     = sQ + NPB*DP;                        // [ROWS][DP]
  float*  sV     = sK + ROWS*DP;                       // [ROWS][DP]
  float*  sAtt   = sV + ROWS*DP;                       // [NPB][MAILS][HEADS]
  float*  sRst   = sAtt + NPB*MAILS*HEADS;             // [NPB][DMODEL]
  float*  sMu    = sRst + NPB*DMODEL;                  // [NPB]
  float*  sRsig  = sMu + NPB;                          // [NPB]

  const int tid   = threadIdx.x;
  const int lane  = tid & 31;
  const int wid   = tid >> 5;          // 0..6, = output column tile
  const int node0 = blockIdx.x * NPB;

  if (tid == 0 && node0 + NPB < n)
    __builtin_prefetch(mem_input + (size_t)(node0 + NPB) * (MAILS*DIN), 0, 1);

  // ---- Phase A: stage mem tile, vectorized float4 -> packed bf16 ----
  for (int idx = tid; idx < NPB*(DMODEL/4); idx += THREADS) {     // 16*25
    int r = idx / 25, q = idx - r*25;
    int ng = node0 + r; if (ng > n - 1) ng = n - 1;
    f32x4 v = *(const f32x4*)(mem + (size_t)ng*DMODEL + q*4);
    u32x2 p; p.x = pack2bf(v.x, v.y); p.y = pack2bf(v.z, v.w);
    *(u32x2*)(sA + r*KQ + q*4) = p;
  }
  for (int idx = tid; idx < NPB*((KQ-DMODEL)/2); idx += THREADS) { // zero pad 100..127
    int r = idx / 14, q = idx - r*14;
    *(unsigned*)(sA + r*KQ + DMODEL + q*2) = 0u;
  }
  // ---- stage mails, vectorized float4 -> packed bf16 ----
  for (int idx = tid; idx < ROWS*(DIN/4); idx += THREADS) {       // 160*25
    int r = idx / 25, q = idx - r*25;
    int nl = r / MAILS, m = r - nl*MAILS;
    int ng = node0 + nl; if (ng > n - 1) ng = n - 1;
    f32x4 v = *(const f32x4*)(mem_input + (size_t)ng*(MAILS*DIN) + m*DIN + q*4);
    u32x2 p; p.x = pack2bf(v.x, v.y); p.y = pack2bf(v.z, v.w);
    *(u32x2*)(sMails + r*KKV + q*4) = p;
  }
  // ---- time encodings: two cos per iteration, packed store ----
  for (int idx = tid; idx < ROWS*(DTENC/2); idx += THREADS) {     // 160*50
    int r = idx / 50, q = idx - r*50;
    int e = q*2;
    int nl = r / MAILS, m = r - nl*MAILS;
    int ng = node0 + nl; if (ng > n - 1) ng = n - 1;
    float dt = ts[ng] - mail_ts[ng*MAILS + m];
    f32x2 w = *(const f32x2*)(te_w + e);
    f32x2 b = *(const f32x2*)(te_b + e);
    float c0 = __cosf(dt * w.x + b.x);
    float c1 = __cosf(dt * w.y + b.y);
    *(unsigned*)(sMails + r*KKV + DIN + e) = pack2bf(c0, c1);
  }
  for (int idx = tid; idx < ROWS*((KKV-200)/2); idx += THREADS) { // zero pad 200..223
    int r = idx / 12, q = idx - r*12;
    *(unsigned*)(sMails + r*KKV + 200 + q*2) = 0u;
  }
  __syncthreads();

  // ---- Phase B: WMMA projections; wave w handles output cols [w*16, w*16+16) ----
  {
    const int col = wid*16 + (lane & 15);
    const int hl  = lane >> 4;

    // Q = memTile @ wq + bq  -> sQ
    {
      v8f acc = {};
      const __bf16* wb = ws + WS_WQT + (size_t)col * KQ + hl*16;
#pragma unroll
      for (int k = 0; k < KQ/32; ++k) {
        v16bf a = load_a_lds(sA + k*32, KQ, lane);
        v16bf b = load_b16(wb + k*32);
        acc = wmma_bf16(a, b, acc);
      }
      float bias = (col < DMODEL) ? bq[col] : 0.f;
#pragma unroll
      for (int r = 0; r < 8; ++r)
        sQ[(r + hl*8)*DP + col] = acc[r] + bias;
    }
    // K = mails @ wk + bk  (hoist the 7 B fragments, sweep 10 row tiles)
    {
      v16bf bfrag[KKV/32];
      const __bf16* wb = ws + WS_WKT + (size_t)col * KKV + hl*16;
#pragma unroll
      for (int k = 0; k < KKV/32; ++k) bfrag[k] = load_b16(wb + k*32);
      float bias = (col < DMODEL) ? bk[col] : 0.f;
      for (int rt = 0; rt < ROWS/16; ++rt) {
        v8f acc = {};
#pragma unroll
        for (int k = 0; k < KKV/32; ++k) {
          v16bf a = load_a_lds(sMails + rt*16*KKV + k*32, KKV, lane);
          acc = wmma_bf16(a, bfrag[k], acc);
        }
#pragma unroll
        for (int r = 0; r < 8; ++r)
          sK[(rt*16 + r + hl*8)*DP + col] = acc[r] + bias;
      }
    }
    // V = mails @ wv + bv
    {
      v16bf bfrag[KKV/32];
      const __bf16* wb = ws + WS_WVT + (size_t)col * KKV + hl*16;
#pragma unroll
      for (int k = 0; k < KKV/32; ++k) bfrag[k] = load_b16(wb + k*32);
      float bias = (col < DMODEL) ? bv[col] : 0.f;
      for (int rt = 0; rt < ROWS/16; ++rt) {
        v8f acc = {};
#pragma unroll
        for (int k = 0; k < KKV/32; ++k) {
          v16bf a = load_a_lds(sMails + rt*16*KKV + k*32, KKV, lane);
          acc = wmma_bf16(a, bfrag[k], acc);
        }
#pragma unroll
        for (int r = 0; r < 8; ++r)
          sV[(rt*16 + r + hl*8)*DP + col] = acc[r] + bias;
      }
    }
  }
  __syncthreads();

  // ---- Phase C: attention (leaky-relu logits, softmax over mailbox) ----
  if (tid < NPB*HEADS) {                 // 32 threads: one (node, head) each
    int nl = tid >> 1, h = tid & 1;
    int base = h * (DMODEL/HEADS);       // 0 or 50
    float logit[MAILS];
    float mx = -1e30f;
    for (int m = 0; m < MAILS; ++m) {
      const float* qp = sQ + nl*DP + base;
      const float* kp = sK + (nl*MAILS + m)*DP + base;
      float s = 0.f;
      for (int d = 0; d < DMODEL/HEADS; ++d) s += qp[d]*kp[d];
      s = (s > 0.f) ? s : 0.2f*s;        // leaky_relu(0.2)
      logit[m] = s;
      mx = fmaxf(mx, s);
    }
    float sum = 0.f;
    for (int m = 0; m < MAILS; ++m) { logit[m] = __expf(logit[m]-mx); sum += logit[m]; }
    float inv = 1.f / sum;
    for (int m = 0; m < MAILS; ++m) sAtt[(nl*MAILS + m)*HEADS + h] = logit[m]*inv;
  }
  __syncthreads();

  // weighted V sum + residual
  for (int idx = tid; idx < NPB*DMODEL; idx += THREADS) {
    int nl = idx / DMODEL, d = idx - nl*DMODEL;
    int h = d / (DMODEL/HEADS);
    float acc = 0.f;
    for (int m = 0; m < MAILS; ++m)
      acc += sAtt[(nl*MAILS + m)*HEADS + h] * sV[(nl*MAILS + m)*DP + d];
    int ng = node0 + nl; if (ng > n - 1) ng = n - 1;
    sRst[idx] = acc + mem[(size_t)ng*DMODEL + d];
  }
  __syncthreads();

  // LayerNorm stats (one thread per node)
  if (tid < NPB) {
    float mu = 0.f;
    for (int d = 0; d < DMODEL; ++d) mu += sRst[tid*DMODEL + d];
    mu *= (1.f/DMODEL);
    float var = 0.f;
    for (int d = 0; d < DMODEL; ++d) {
      float x = sRst[tid*DMODEL + d] - mu; var += x*x;
    }
    var *= (1.f/DMODEL);
    sMu[tid]   = mu;
    sRsig[tid] = rsqrtf(var + 1e-5f);
  }
  __syncthreads();

  // normalize -> bf16 MLP A tile (reuse sA, zero-padded, packed stores)
  for (int idx = tid; idx < NPB*(KQ/2); idx += THREADS) {   // 16*64
    int nl = idx >> 6, q = idx & 63;
    int c = q*2;
    unsigned pk = 0u;
    if (c < DMODEL) {
      float mu = sMu[nl], rs = sRsig[nl];
      float a = (sRst[nl*DMODEL + c]   - mu) * rs * ln_g[c]   + ln_b[c];
      float b = (sRst[nl*DMODEL + c+1] - mu) * rs * ln_g[c+1] + ln_b[c+1];
      pk = pack2bf(a, b);
    }
    *(unsigned*)(sA + nl*KQ + c) = pk;
  }
  __syncthreads();

  // ---- Phase D: MLP via WMMA, ReLU, store f32 output ----
  {
    const int col = wid*16 + (lane & 15);
    const int hl  = lane >> 4;
    v8f acc = {};
    const __bf16* wb = ws + WS_MLPT + (size_t)col * KQ + hl*16;
#pragma unroll
    for (int k = 0; k < KQ/32; ++k) {
      v16bf a = load_a_lds(sA + k*32, KQ, lane);
      v16bf b = load_b16(wb + k*32);
      acc = wmma_bf16(a, b, acc);
    }
    if (col < DMODEL) {
      float bias = mlp_b[col];
#pragma unroll
      for (int r = 0; r < 8; ++r) {
        int ng = node0 + r + hl*8;
        if (ng < n) out[(size_t)ng*DMODEL + col] = fmaxf(acc[r] + bias, 0.f);
      }
    }
  }
}

// ---------------- host entry ----------------
extern "C" void kernel_launch(void* const* d_in, const int* in_sizes, int n_in,
                              void* d_out, int out_size, void* d_ws, size_t ws_size,
                              hipStream_t stream) {
  const float* mem       = (const float*)d_in[0];
  const float* mem_input = (const float*)d_in[1];
  const float* ts        = (const float*)d_in[2];
  const float* mail_ts   = (const float*)d_in[3];
  const float* wq        = (const float*)d_in[4];
  const float* bq        = (const float*)d_in[5];
  const float* wk        = (const float*)d_in[6];
  const float* bk        = (const float*)d_in[7];
  const float* wv        = (const float*)d_in[8];
  const float* bv        = (const float*)d_in[9];
  const float* ln_g      = (const float*)d_in[10];
  const float* ln_b      = (const float*)d_in[11];
  const float* mlp_w     = (const float*)d_in[12];
  const float* mlp_b     = (const float*)d_in[13];
  const float* te_w      = (const float*)d_in[14];
  const float* te_b      = (const float*)d_in[15];
  float*  out = (float*)d_out;
  __bf16* ws  = (__bf16*)d_ws;

  const int n = in_sizes[0] / DMODEL;

  prep_weights_kernel<<<(WS_TOTAL + 255)/256, 256, 0, stream>>>(wq, wk, wv, mlp_w, ws);

  (void)hipFuncSetAttribute((const void*)tmu_fused_kernel,
                            hipFuncAttributeMaxDynamicSharedMemorySize,
                            (int)SMEM_BYTES);

  const int blocks = (n + NPB - 1) / NPB;
  tmu_fused_kernel<<<blocks, THREADS, SMEM_BYTES, stream>>>(
      mem, mem_input, ts, mail_ts, bq, bk, bv, ln_g, ln_b, mlp_b,
      te_w, te_b, ws, out, n);
}
